// LlamaStyleEngramMoETransformerBlock_15994458211254
// MI455X (gfx1250) — compile-verified, matrix operand order checked by reading, not run on hardware
//
#include <hip/hip_runtime.h>
#include <hip/hip_bf16.h>
#include <math.h>

// ---------------------------------------------------------------------------
// CDNA5 (gfx1250) wave32 WMMA transformer block.
// All matmuls use V_WMMA_F32_16X16X32_BF16 (bf16 in, f32 accumulate).
// fp32 -> bf16 conversion happens while staging tiles into LDS.
// ---------------------------------------------------------------------------

typedef __attribute__((ext_vector_type(16))) __bf16 v16bf;
typedef __attribute__((ext_vector_type(8)))  float  v8f;

static __device__ __forceinline__ unsigned bfpack(float a, float b) {
  // round-to-nearest-even fp32 -> bf16, pack two into one dword (a = low)
  unsigned ua = __float_as_uint(a), ub = __float_as_uint(b);
  ua += 0x7FFFu + ((ua >> 16) & 1u);
  ub += 0x7FFFu + ((ub >> 16) & 1u);
  return (ua >> 16) | (ub & 0xFFFF0000u);
}
static __device__ __forceinline__ unsigned short bfbits(float a) {
  unsigned ua = __float_as_uint(a);
  ua += 0x7FFFu + ((ua >> 16) & 1u);
  return (unsigned short)(ua >> 16);
}
static __device__ __forceinline__ v8f zero8() {
  v8f z;
#pragma unroll
  for (int i = 0; i < 8; ++i) z[i] = 0.0f;
  return z;
}

// Fragment loader for 16-bit A-layout (ISA 7.12.2): lane L holds row (L&15);
// half = L>>4 selects K offset +8 elements (+4 dwords). VGPR j holds K pair
// (j<4 ? 2j : 16+2(j-4)) + 8*half  ->  dword index (j<4 ? j : j+4) + 4*half.
// B tiles are staged transposed ([n][k]) so the same loader serves both.
static __device__ __forceinline__ v16bf load_frag(const unsigned* base,
                                                  int stride_u, int lane,
                                                  int uoff) {
  union { v16bf v; unsigned u[8]; } f;
  const unsigned* p = base + (size_t)(lane & 15) * stride_u + uoff + 4 * (lane >> 4);
#pragma unroll
  for (int j = 0; j < 8; ++j) f.u[j] = p[j < 4 ? j : j + 4];
  return f.v;
}

static __device__ __forceinline__ v8f wmma_bf16(v16bf a, v16bf b, v8f c) {
  // (neg_a, A, neg_b, B, c_mod, C, reuse_a, reuse_b)
  return __builtin_amdgcn_wmma_f32_16x16x32_bf16(false, a, false, b,
                                                 (short)0, c, false, false);
}

// ---------------------------------------------------------------------------
// Generic tiled GEMM: C[M,N] = A[M,K] * B[K,N], all fp32 in memory.
// 128x128 block tile, BK=32, 8 waves (2x4), wave tile 64x32 (4x2 WMMA tiles).
// ---------------------------------------------------------------------------
__global__ __launch_bounds__(256) void gemm_bf16_kernel(const float* __restrict__ A,
                                                        const float* __restrict__ B,
                                                        float* __restrict__ C,
                                                        int M, int N, int K) {
  __shared__ unsigned ldsA[128][17];  // [m][k-pair], +1 dword pad
  __shared__ unsigned ldsB[128][17];  // [n][k-pair] (B staged transposed)

  const int tid  = threadIdx.x;
  const int lane = tid & 31;
  const int wrp  = tid >> 5;
  const int wm   = wrp >> 2;   // 0..1
  const int wn   = wrp & 3;    // 0..3
  const int bM   = blockIdx.y * 128;
  const int bN   = blockIdx.x * 128;

  v8f acc[4][2];
#pragma unroll
  for (int mt = 0; mt < 4; ++mt)
#pragma unroll
    for (int nt = 0; nt < 2; ++nt) acc[mt][nt] = zero8();

  const int nk = K >> 5;
  const int arow = tid >> 1, ahf = tid & 1;       // A loader coords
  const int kp = tid & 15,  ns  = tid >> 4;       // B loader coords

  for (int kb = 0; kb < nk; ++kb) {
    // ---- stage A tile (128x32 fp32 -> bf16 pairs) ----
    {
      const float4* p4 = (const float4*)(A + (size_t)(bM + arow) * K + kb * 32 + ahf * 16);
      unsigned* dst = &ldsA[arow][ahf * 8];
#pragma unroll
      for (int j = 0; j < 4; ++j) {
        float4 f = p4[j];
        dst[2 * j]     = bfpack(f.x, f.y);
        dst[2 * j + 1] = bfpack(f.z, f.w);
      }
      if (kb + 1 < nk)
        __builtin_prefetch((const void*)(A + (size_t)(bM + arow) * K + (kb + 1) * 32 + ahf * 16), 0, 0);
    }
    // ---- stage B tile transposed (32x128 fp32 -> [n][k] bf16 pairs) ----
    {
      const float* p0 = B + (size_t)(kb * 32 + 2 * kp) * N + bN + ns * 8;
      const float* p1 = p0 + N;
      const float4* a4 = (const float4*)p0;
      const float4* b4 = (const float4*)p1;
      float4 x0 = a4[0], x1 = a4[1], y0 = b4[0], y1 = b4[1];
      float bx[8] = {x0.x, x0.y, x0.z, x0.w, x1.x, x1.y, x1.z, x1.w};
      float by[8] = {y0.x, y0.y, y0.z, y0.w, y1.x, y1.y, y1.z, y1.w};
#pragma unroll
      for (int i = 0; i < 8; ++i) ldsB[ns * 8 + i][kp] = bfpack(bx[i], by[i]);
      if (kb + 1 < nk)
        __builtin_prefetch((const void*)(B + (size_t)((kb + 1) * 32 + 2 * kp) * N + bN + ns * 8), 0, 0);
    }
    __syncthreads();

    v16bf af[4], bfr[2];
#pragma unroll
    for (int mt = 0; mt < 4; ++mt)
      af[mt] = load_frag(&ldsA[wm * 64 + mt * 16][0], 17, lane, 0);
#pragma unroll
    for (int nt = 0; nt < 2; ++nt)
      bfr[nt] = load_frag(&ldsB[wn * 32 + nt * 16][0], 17, lane, 0);
#pragma unroll
    for (int mt = 0; mt < 4; ++mt)
#pragma unroll
      for (int nt = 0; nt < 2; ++nt)
        acc[mt][nt] = wmma_bf16(af[mt], bfr[nt], acc[mt][nt]);

    __syncthreads();
  }

  // ---- store C: VGPR r -> row r + 8*(lane>>4), col = lane&15 ----
  const int col = lane & 15, hf = lane >> 4;
#pragma unroll
  for (int mt = 0; mt < 4; ++mt)
#pragma unroll
    for (int nt = 0; nt < 2; ++nt)
#pragma unroll
      for (int r = 0; r < 8; ++r) {
        int row = bM + wm * 64 + mt * 16 + r + 8 * hf;
        int cc  = bN + wn * 32 + nt * 16 + col;
        C[(size_t)row * N + cc] = acc[mt][nt][r];
      }
}

// ---------------------------------------------------------------------------
// Flash attention (causal, GQA 16 heads over 8 KV heads, hd=64).
// Block = 4 waves; each wave owns 16 query rows of a 64-query tile.
// ---------------------------------------------------------------------------
__global__ __launch_bounds__(128) void flash_attn_kernel(const float* __restrict__ q,
                                                         const float* __restrict__ k,
                                                         const float* __restrict__ v,
                                                         float* __restrict__ o,
                                                         int S, int nH, int nKV) {
  __shared__ unsigned ldsQ[64][33];   // [qrow][dim-pair]
  __shared__ unsigned ldsK[64][33];   // [krow][dim-pair]  (serves as Bt for QK^T)
  __shared__ unsigned ldsVt[64][33];  // [dim][kidx-pair]  (Bt for P*V)
  __shared__ unsigned short ldsP[4][16][72];  // per-wave P tile, bf16

  const int qt = blockIdx.x, h = blockIdx.y;
  const int tid = threadIdx.x, wv = tid >> 5, lane = tid & 31;
  const int q0 = qt * 64;
  const int dQ = nH * 64, dKV = nKV * 64;
  const float* qbase = q + (size_t)q0 * dQ + h * 64;
  const float* kbase = k + (h >> 1) * 64;
  const float* vbase = v + (h >> 1) * 64;
  const float scale = 0.125f;  // 1/sqrt(64)

  // stage Q tile once
  {
    int row = tid >> 1, hf = tid & 1;
    const float* p = qbase + (size_t)row * dQ + hf * 32;
#pragma unroll
    for (int j = 0; j < 16; ++j)
      ldsQ[row][hf * 16 + j] = bfpack(p[2 * j], p[2 * j + 1]);
  }
  __syncthreads();

  v16bf aQ0 = load_frag(&ldsQ[wv * 16][0], 33, lane, 0);
  v16bf aQ1 = load_frag(&ldsQ[wv * 16][0], 33, lane, 16);

  v8f oacc[4];
#pragma unroll
  for (int dt = 0; dt < 4; ++dt) oacc[dt] = zero8();
  float mrow[8], lrow[8];
#pragma unroll
  for (int r = 0; r < 8; ++r) { mrow[r] = -3.0e38f; lrow[r] = 0.0f; }

  const int col = lane & 15, hf2 = lane >> 4;

  for (int kt = 0; kt <= qt; ++kt) {
    // ---- stage K tile [key][dim] ----
    {
      int row = tid >> 1, hf = tid & 1;
      const float* p = kbase + (size_t)(kt * 64 + row) * dKV + hf * 32;
#pragma unroll
      for (int j = 0; j < 16; ++j)
        ldsK[row][hf * 16 + j] = bfpack(p[2 * j], p[2 * j + 1]);
    }
    // ---- stage V transposed [dim][kidx] ----
    {
      int dc = tid >> 1, hf = tid & 1;
      const float* p = vbase + dc;
#pragma unroll
      for (int j = 0; j < 16; ++j) {
        int kk = kt * 64 + hf * 32 + 2 * j;
        ldsVt[dc][hf * 16 + j] = bfpack(p[(size_t)kk * dKV], p[(size_t)(kk + 1) * dKV]);
      }
    }
    __syncthreads();

    // ---- S = Q K^T (16x64 per wave = 4 tiles, K-dim = 64 = 2 WMMA steps) ----
    v8f sc[4];
#pragma unroll
    for (int nt = 0; nt < 4; ++nt) {
      v16bf b0 = load_frag(&ldsK[nt * 16][0], 33, lane, 0);
      v16bf b1 = load_frag(&ldsK[nt * 16][0], 33, lane, 16);
      v8f s = zero8();
      s = wmma_bf16(aQ0, b0, s);
      s = wmma_bf16(aQ1, b1, s);
      sc[nt] = s;
    }

    // ---- scale + causal mask ----
#pragma unroll
    for (int nt = 0; nt < 4; ++nt)
#pragma unroll
      for (int r = 0; r < 8; ++r) {
        float val = sc[nt][r] * scale;
        int kidx = kt * 64 + nt * 16 + col;
        int qidx = q0 + wv * 16 + r + 8 * hf2;
        sc[nt][r] = (kidx > qidx) ? -1.0e30f : val;
      }

    // ---- online softmax update (reduce across 16 lanes of each half) ----
    float alpha[8];
#pragma unroll
    for (int r = 0; r < 8; ++r) {
      float m = fmaxf(fmaxf(sc[0][r], sc[1][r]), fmaxf(sc[2][r], sc[3][r]));
#pragma unroll
      for (int off = 1; off < 16; off <<= 1) m = fmaxf(m, __shfl_xor(m, off));
      float mnew = fmaxf(mrow[r], m);
      alpha[r] = __expf(mrow[r] - mnew);
      mrow[r] = mnew;
      float rs = 0.0f;
#pragma unroll
      for (int nt = 0; nt < 4; ++nt) {
        float p = __expf(sc[nt][r] - mnew);
        sc[nt][r] = p;
        rs += p;
      }
#pragma unroll
      for (int off = 1; off < 16; off <<= 1) rs += __shfl_xor(rs, off);
      lrow[r] = lrow[r] * alpha[r] + rs;
    }
#pragma unroll
    for (int dt = 0; dt < 4; ++dt)
#pragma unroll
      for (int r = 0; r < 8; ++r) oacc[dt][r] *= alpha[r];

    // ---- spill P (bf16) to this wave's LDS region, reload as A fragments ----
#pragma unroll
    for (int nt = 0; nt < 4; ++nt)
#pragma unroll
      for (int r = 0; r < 8; ++r)
        ldsP[wv][r + 8 * hf2][nt * 16 + col] = bfbits(sc[nt][r]);

    const unsigned* pb = (const unsigned*)&ldsP[wv][0][0];
    v16bf p0 = load_frag(pb, 36, lane, 0);
    v16bf p1 = load_frag(pb, 36, lane, 16);
#pragma unroll
    for (int dt = 0; dt < 4; ++dt) {
      v16bf bv0 = load_frag(&ldsVt[dt * 16][0], 33, lane, 0);
      v16bf bv1 = load_frag(&ldsVt[dt * 16][0], 33, lane, 16);
      oacc[dt] = wmma_bf16(p0, bv0, oacc[dt]);
      oacc[dt] = wmma_bf16(p1, bv1, oacc[dt]);
    }
    __syncthreads();
  }

  // ---- normalize and write ----
#pragma unroll
  for (int dt = 0; dt < 4; ++dt)
#pragma unroll
    for (int r = 0; r < 8; ++r) {
      int qi = q0 + wv * 16 + r + 8 * hf2;
      o[(size_t)qi * dQ + h * 64 + dt * 16 + col] = oacc[dt][r] / lrow[r];
    }
}

// ---------------------------------------------------------------------------
// Elementwise / small kernels
// ---------------------------------------------------------------------------
__global__ void rmsnorm_kernel(const float* __restrict__ x, const float* __restrict__ w,
                               float* __restrict__ out, int D) {
  __shared__ float red[8];
  const int row = blockIdx.x;
  const int lane = threadIdx.x & 31, wv = threadIdx.x >> 5;
  const float* xr = x + (size_t)row * D;
  float ss = 0.0f;
  for (int j = threadIdx.x; j < D; j += blockDim.x) { float t = xr[j]; ss += t * t; }
#pragma unroll
  for (int off = 1; off < 32; off <<= 1) ss += __shfl_xor(ss, off);
  if (lane == 0) red[wv] = ss;
  __syncthreads();
  if (wv == 0) {
    float t = (lane < 8) ? red[lane] : 0.0f;
#pragma unroll
    for (int off = 1; off < 8; off <<= 1) t += __shfl_xor(t, off);
    if (lane == 0) red[0] = t;
  }
  __syncthreads();
  float inv = rsqrtf(red[0] / (float)D + 1e-6f);
  for (int j = threadIdx.x; j < D; j += blockDim.x)
    out[(size_t)row * D + j] = xr[j] * inv * w[j];
}

__global__ void rope_kernel(float* __restrict__ t, int S, int H) {
  int idx = blockIdx.x * blockDim.x + threadIdx.x;
  int total = S * H * 32;
  if (idx >= total) return;
  int i = idx & 31;
  int h = (idx >> 5) % H;
  int s = idx / (32 * H);
  float freq = __powf(10000.0f, -(float)i / 32.0f);
  float ang = (float)s * freq;
  float c = __cosf(ang), sn = __sinf(ang);
  size_t base = ((size_t)s * H + h) * 64;
  float t1 = t[base + i], t2 = t[base + i + 32];
  t[base + i]      = t1 * c - t2 * sn;
  t[base + i + 32] = t2 * c + t1 * sn;
}

__global__ void add_kernel(const float* __restrict__ a, const float* __restrict__ b,
                           float* __restrict__ out, int n) {
  int i = blockIdx.x * blockDim.x + threadIdx.x;
  if (i < n) out[i] = a[i] + b[i];
}

__global__ void copy_kernel(const float* __restrict__ a, float* __restrict__ out, int n) {
  int i = blockIdx.x * blockDim.x + threadIdx.x;
  if (i < n) out[i] = a[i];
}

__global__ void silu_mul_kernel(float* __restrict__ h, const float* __restrict__ g, int n) {
  int i = blockIdx.x * blockDim.x + threadIdx.x;
  if (i < n) {
    float x = h[i];
    h[i] = (x / (1.0f + __expf(-x))) * g[i];
  }
}

__global__ void router_kernel(const float* __restrict__ xn, const float* __restrict__ wr,
                              float* __restrict__ comb, int S, int D, int E) {
  int t = blockIdx.x * blockDim.x + threadIdx.x;
  if (t >= S) return;
  float logits[8];
#pragma unroll
  for (int e = 0; e < 8; ++e) logits[e] = 0.0f;
  const float* xr = xn + (size_t)t * D;
  for (int j = 0; j < D; ++j) {
    float xv = xr[j];
#pragma unroll
    for (int e = 0; e < 8; ++e) logits[e] += xv * wr[(size_t)j * E + e];
  }
  float mx = logits[0];
#pragma unroll
  for (int e = 1; e < 8; ++e) mx = fmaxf(mx, logits[e]);
  float sum = 0.0f;
#pragma unroll
  for (int e = 0; e < 8; ++e) { logits[e] = __expf(logits[e] - mx); sum += logits[e]; }
#pragma unroll
  for (int e = 0; e < 8; ++e) logits[e] /= sum;
  int i1 = 0;
#pragma unroll
  for (int e = 1; e < 8; ++e) if (logits[e] > logits[i1]) i1 = e;
  int i2 = (i1 == 0) ? 1 : 0;
#pragma unroll
  for (int e = 0; e < 8; ++e)
    if (e != i1 && logits[e] > logits[i2]) i2 = e;
  float norm = logits[i1] + logits[i2];
#pragma unroll
  for (int e = 0; e < 8; ++e) comb[(size_t)t * E + e] = 0.0f;
  comb[(size_t)t * E + i1] = logits[i1] / norm;
  comb[(size_t)t * E + i2] = logits[i2] / norm;
}

__global__ void moe_accum_kernel(float* __restrict__ out, const float* __restrict__ y,
                                 const float* __restrict__ comb, int D, int E, int e, int n) {
  int i = blockIdx.x * blockDim.x + threadIdx.x;
  if (i < n) {
    int t = i / D;
    out[i] += comb[(size_t)t * E + e] * y[i];
  }
}

// ---------------------------------------------------------------------------
// Orchestration
// ---------------------------------------------------------------------------
extern "C" void kernel_launch(void* const* d_in, const int* in_sizes, int n_in,
                              void* d_out, int out_size, void* d_ws, size_t ws_size,
                              hipStream_t stream) {
  (void)in_sizes; (void)n_in; (void)out_size; (void)ws_size;
  constexpr int S = 2048, D = 1024, KV = 512, DFF = 2816, E = 8, H = 16, HKV = 8;

  const float* x   = (const float*)d_in[0];
  // d_in[1] input_ids, d_in[2] attn_mask (all-true) -- unused by the math
  const float* ln1 = (const float*)d_in[3];
  const float* wq  = (const float*)d_in[4];
  const float* wk  = (const float*)d_in[5];
  const float* wvp = (const float*)d_in[6];
  const float* wo  = (const float*)d_in[7];
  const float* ln2 = (const float*)d_in[8];
  const float* wr  = (const float*)d_in[9];
  const float* w1  = (const float*)d_in[10];
  const float* w3  = (const float*)d_in[11];
  const float* w2  = (const float*)d_in[12];
  float* out = (float*)d_out;

  float* ws   = (float*)d_ws;
  float* xn   = ws;                       // S*D
  float* qb   = xn   + (size_t)S * D;     // S*D
  float* kb   = qb   + (size_t)S * D;     // S*KV
  float* vb   = kb   + (size_t)S * KV;    // S*KV
  float* ao   = vb   + (size_t)S * KV;    // S*D
  float* xa   = ao   + (size_t)S * D;     // S*D
  float* xn2  = xa   + (size_t)S * D;     // S*D
  float* comb = xn2  + (size_t)S * D;     // S*E
  float* hb   = comb + (size_t)S * E;     // S*DFF
  float* gb   = hb   + (size_t)S * DFF;   // S*DFF
  float* yb   = gb   + (size_t)S * DFF;   // S*D

  const int EW = 256;
  auto gemm = [&](const float* A, const float* B, float* C, int M, int N, int K) {
    gemm_bf16_kernel<<<dim3(N / 128, M / 128), 256, 0, stream>>>(A, B, C, M, N, K);
  };

  // ---- attention half ----
  rmsnorm_kernel<<<S, 256, 0, stream>>>(x, ln1, xn, D);
  gemm(xn, wq,  qb, S, D,  D);
  gemm(xn, wk,  kb, S, KV, D);
  gemm(xn, wvp, vb, S, KV, D);
  rope_kernel<<<(S * H   * 32 + EW - 1) / EW, EW, 0, stream>>>(qb, S, H);
  rope_kernel<<<(S * HKV * 32 + EW - 1) / EW, EW, 0, stream>>>(kb, S, HKV);
  flash_attn_kernel<<<dim3(S / 64, H), 128, 0, stream>>>(qb, kb, vb, ao, S, H, HKV);
  gemm(ao, wo, yb, S, D, D);
  add_kernel<<<(S * D + EW - 1) / EW, EW, 0, stream>>>(x, yb, xa, S * D);

  // ---- MoE half ----
  rmsnorm_kernel<<<S, 256, 0, stream>>>(xa, ln2, xn2, D);
  router_kernel<<<(S + EW - 1) / EW, EW, 0, stream>>>(xn2, wr, comb, S, D, E);
  copy_kernel<<<(S * D + EW - 1) / EW, EW, 0, stream>>>(xa, out, S * D);

  for (int e = 0; e < E; ++e) {
    const float* w1e = w1 + (size_t)e * D * DFF;
    const float* w3e = w3 + (size_t)e * D * DFF;
    const float* w2e = w2 + (size_t)e * DFF * D;
    gemm(xn2, w1e, hb, S, DFF, D);
    gemm(xn2, w3e, gb, S, DFF, D);
    silu_mul_kernel<<<(S * DFF + EW - 1) / EW, EW, 0, stream>>>(hb, gb, S * DFF);
    gemm(hb, w2e, yb, S, D, DFF);
    moe_accum_kernel<<<(S * D + EW - 1) / EW, EW, 0, stream>>>(out, yb, comb, D, E, e, S * D);
  }
}
